// NERLSTM_CRF_20091857011457
// MI455X (gfx1250) — compile-verified, matrix operand order checked by reading, not run on hardware
//
#include <hip/hip_runtime.h>

// ---------------------------------------------------------------------------
// BiLSTM-CRF for MI455X (gfx1250, wave32, WMMA bf16 16x16x32)
//   V=30000 E=256 H=256 (per dir) 4H=1024  B=64 T=512 K=9
// Gx and hs are kept in C-fragment-major layout so the sequential LSTM
// kernel does only contiguous 32B/lane vector loads/stores per step.
// ---------------------------------------------------------------------------

#define Tn   512
#define Bn   64
#define En   256
#define Hn   256
#define Kn   9
#define NG   1024          // 4*H gate width per direction
#define HLDS 258           // padded LDS row stride in halves (bank-spread)
#define GXT  65536         // floats per (dir,t) slab: 32 waves * 4 g * 2 rt * 32 lanes * 8

typedef __attribute__((ext_vector_type(16))) __bf16 v16bf;
typedef __attribute__((ext_vector_type(8)))  float  v8f;

union F8      { v8f   v; float          f[8];  };
union BF16x16 { v16bf v; unsigned short u[16]; };

__device__ __forceinline__ unsigned short f32_to_bf16(float x) {
  unsigned int u = __float_as_uint(x);
  u += 0x7FFFu + ((u >> 16) & 1u);              // round-to-nearest-even
  return (unsigned short)(u >> 16);
}
__device__ __forceinline__ float sigmoidf(float x) { return 1.0f / (1.0f + __expf(-x)); }

__device__ __forceinline__ v8f wmma_bf16(v16bf a, v16bf b, v8f c) {
  // D(16x16,f32) = A(16x32,bf16) x B(32x16,bf16) + C
  return __builtin_amdgcn_wmma_f32_16x16x32_bf16(false, a, false, b, (short)0, c,
                                                 false, false);
}

// A fragment (16x32 bf16) from LDS laid out [row][k], row stride HLDS halves.
__device__ __forceinline__ v16bf load_fragA_lds(const unsigned short* hlds,
                                                int row0, int kt, int lane) {
  const int m   = row0 + (lane & 15);
  const int klo = kt * 32 + ((lane & 16) ? 8 : 0);
  const unsigned short* p = hlds + m * HLDS + klo;
  BF16x16 r;
#pragma unroll
  for (int e = 0; e < 8; ++e) { r.u[e] = p[e]; r.u[8 + e] = p[16 + e]; }
  return r.v;
}

// B fragment (32x16 bf16) from the pre-swizzled fragment buffer:
// layout [kt][nt][lane][16 halves] -> one coalesced 32B load per lane.
__device__ __forceinline__ v16bf load_fragB(const unsigned short* base,
                                            int kt, int nt, int lane) {
  const unsigned short* p = base + ((((kt * 64 + nt) * 32) + lane) << 4);
  return *(const v16bf*)p;
}

// ---------------------------------------------------------------------------
// Kernel 1: pre-swizzle the 4 weight matrices [1024][256] f32 into bf16
// B-fragment-major layout. buf order: Wih_f, Wih_b, Whh_f, Whh_b.
// ---------------------------------------------------------------------------
__global__ __launch_bounds__(256)
void convert_weights_kernel(const float* __restrict__ Wih_f,
                            const float* __restrict__ Wih_b,
                            const float* __restrict__ Whh_f,
                            const float* __restrict__ Whh_b,
                            unsigned short* __restrict__ frag) {
  int idx = blockIdx.x * blockDim.x + threadIdx.x;
  if (idx >= 4 * 262144) return;
  const int buf  = idx >> 18;
  const int rem  = idx & 262143;                 // kt*32768 + nt*512 + lane*16 + e
  const int kt   = rem >> 15;
  const int nt   = (rem >> 9) & 63;
  const int lane = (rem >> 4) & 31;
  const int e    = rem & 15;
  const int n    = nt * 16 + (lane & 15);
  const int kin  = ((e < 8) ? e : (8 + e)) + ((lane & 16) ? 8 : 0);
  const int k    = kt * 32 + kin;
  const float* src = (buf == 0) ? Wih_f : (buf == 1) ? Wih_b
                   : (buf == 2) ? Whh_f : Whh_b;
  frag[idx] = f32_to_bf16(src[n * 256 + k]);     // B[k][n] = W[n][k]
}

__global__ __launch_bounds__(256)
void bias_kernel(const float* bih_f, const float* bhh_f,
                 const float* bih_b, const float* bhh_b, float* bsum) {
  int i = blockIdx.x * blockDim.x + threadIdx.x;
  if (i < NG)            bsum[i] = bih_f[i] + bhh_f[i];
  else if (i < 2 * NG)   bsum[i] = bih_b[i - NG] + bhh_b[i - NG];
}

// ---------------------------------------------------------------------------
// Kernel 2: embedding gather + input projection GEMM.
//   M = 32768 (= T slabs of 64 batch rows), N = 1024/dir, K = 256.
//   One 64x64 tile per WG (rowg == t).  Output written in the LSTM kernel's
//   C-fragment-major layout: slab(dir,t) + ((wave*4+g)*2+rt)*256 + lane*8 + v.
// ---------------------------------------------------------------------------
__global__ __launch_bounds__(256)
void input_gemm_kernel(const int* __restrict__ x, const float* __restrict__ embed,
                       const unsigned short* __restrict__ wih_frag,
                       const float* __restrict__ bsum, float* __restrict__ Gx) {
  __shared__ int tok[64];
  __shared__ unsigned short alds[64 * HLDS];
  const int tid  = threadIdx.x;
  const int dir  = blockIdx.z;
  const int rowg = blockIdx.y;                   // == t (64 batch rows per slab)
  const int colg = blockIdx.x;                   // 0..15 (64 cols each)
  if (tid < 64) {
    int r = rowg * 64 + tid;                     // r = t*64 + b
    tok[tid] = x[(r & 63) * Tn + (r >> 6)];
  }
  __syncthreads();
  for (int i = tid; i < 64 * 256; i += 256) {
    int r = i >> 8, k = i & 255;
    alds[r * HLDS + k] = f32_to_bf16(embed[(size_t)tok[r] * En + k]);
  }
  __syncthreads();
  const int wave = tid >> 5, lane = tid & 31;
  const int rt_ig = wave & 3;                    // row tile 0..3 (16 rows each)
  const int ch    = wave >> 2;                   // col-tile half 0..1
  const unsigned short* wf = wih_frag + (size_t)dir * 262144;
  F8 acc[2];
  for (int j = 0; j < 2; ++j)
    for (int v = 0; v < 8; ++v) acc[j].f[v] = 0.0f;
#pragma unroll
  for (int kt = 0; kt < 8; ++kt) {
    v16bf a = load_fragA_lds(alds, rt_ig * 16, kt, lane);
#pragma unroll
    for (int j = 0; j < 2; ++j) {
      v16bf b = load_fragB(wf, kt, colg * 4 + ch * 2 + j, lane);
      acc[j].v = wmma_bf16(a, b, acc[j].v);
    }
  }
  // store in LSTM C-fragment-major layout (lane maps to lane, v to v)
  float* slab = Gx + ((size_t)dir * Tn + rowg) * GXT;
#pragma unroll
  for (int j = 0; j < 2; ++j) {
    const int ctg   = colg * 4 + ch * 2 + j;     // global col tile 0..63
    const int g     = ctg >> 4;                  // gate 0..3
    const int jt_l  = ctg & 15;                  // lstm jt
    const int wv_l  = jt_l * 2 + (rt_ig >> 1);   // lstm wave
    const int rt_l  = rt_ig & 1;                 // lstm row-tile within wave
    const float bias = bsum[dir * NG + ctg * 16 + (lane & 15)];
    float* p = slab + ((size_t)(wv_l * 4 + g) * 2 + rt_l) * 256 + (size_t)lane * 8;
#pragma unroll
    for (int v = 0; v < 8; ++v) p[v] = acc[j].f[v] + bias;
  }
}

// ---------------------------------------------------------------------------
// Kernel 3: persistent LSTM recurrence. One 32-wave WG per direction.
//   h (bf16) in LDS, c in registers, Whh fragments streamed from L2.
//   Per step per wave: 8 vector Gx loads, 64 WMMAs, 2 vector hs stores.
// ---------------------------------------------------------------------------
__global__ __launch_bounds__(1024)
void lstm_kernel(const float* __restrict__ Gx,
                 const unsigned short* __restrict__ whh_frag,
                 float* __restrict__ hs) {
  __shared__ unsigned short hlds[64 * HLDS];     // h state, bf16 [64][256]
  const int tid  = threadIdx.x;
  const int dir  = blockIdx.x;
  const int wave = tid >> 5, lane = tid & 31;
  const int jt = wave >> 1, rh = wave & 1;
  const int hl = (lane & 16) ? 8 : 0;
  const int n  = jt * 16 + (lane & 15);          // hidden index this lane owns
  const unsigned short* wf = whh_frag + (size_t)dir * 262144;
  const float* gxd = Gx + (size_t)dir * Tn * GXT;

  for (int i = tid; i < 64 * HLDS; i += 1024) hlds[i] = 0;
  F8 c[2];
  for (int rt = 0; rt < 2; ++rt)
    for (int v = 0; v < 8; ++v) c[rt].f[v] = 0.0f;
  __syncthreads();

  for (int s = 0; s < Tn; ++s) {
    const int tt = dir ? (Tn - 1 - s) : s;
    // C init = Gx (+bias): 8 fragments, each one contiguous 32B/lane load
    const float* gxt = gxd + (size_t)tt * GXT + (size_t)wave * 2048 + (size_t)lane * 8;
    F8 acc[4][2];
#pragma unroll
    for (int g = 0; g < 4; ++g)
#pragma unroll
      for (int rt = 0; rt < 2; ++rt) {
        const float* p = gxt + (g * 2 + rt) * 256;
#pragma unroll
        for (int v = 0; v < 8; ++v) acc[g][rt].f[v] = p[v];
      }
    // 64 WMMAs: G += h @ Whh^T for this wave's 4-gate x 2-rowtile patch
#pragma unroll
    for (int kt = 0; kt < 8; ++kt) {
      v16bf a0 = load_fragA_lds(hlds, rh * 32,      kt, lane);
      v16bf a1 = load_fragA_lds(hlds, rh * 32 + 16, kt, lane);
#pragma unroll
      for (int g = 0; g < 4; ++g) {
        v16bf b = load_fragB(wf, kt, g * 16 + jt, lane);
        acc[g][0].v = wmma_bf16(a0, b, acc[g][0].v);
        acc[g][1].v = wmma_bf16(a1, b, acc[g][1].v);
      }
    }
    if (s + 1 < Tn) {                            // warm L2/L0 for next Gx slab
      const int ttn = dir ? (Tn - 2 - s) : (s + 1);
      __builtin_prefetch(gxd + (size_t)ttn * GXT + (size_t)wave * 2048 + (size_t)lane * 8, 0, 1);
    }
    __syncthreads();                             // all waves done reading h
#pragma unroll
    for (int rt = 0; rt < 2; ++rt) {
      float* hp = hs + ((((size_t)tt * 2 + dir) * 32 + wave) * 2 + rt) * 256
                     + (size_t)lane * 8;
#pragma unroll
      for (int v = 0; v < 8; ++v) {
        const int m  = rh * 32 + rt * 16 + v + hl;   // batch index
        const float ig = sigmoidf(acc[0][rt].f[v]);
        const float fg = sigmoidf(acc[1][rt].f[v]);
        const float gg = tanhf(acc[2][rt].f[v]);
        const float og = sigmoidf(acc[3][rt].f[v]);
        const float cn = fg * c[rt].f[v] + ig * gg;
        c[rt].f[v] = cn;
        const float h = og * tanhf(cn);
        hlds[m * HLDS + n] = f32_to_bf16(h);
        hp[v] = h;                               // contiguous 32B/lane store
      }
    }
    __syncthreads();                             // h published before next read
  }
}

// ---------------------------------------------------------------------------
// Kernel 4: emissions. hs is fragment-major:
//   hs[(((t*2+dir)*32 + wave)*2 + rt)*256 + lane*8 + v]
//   with wave = jt*2 + (b>>5), rt = (b>>4)&1, lane = (c&15) + ((b>>3)&1)*16,
//   v = b&7, jt = (c&255)>>4, dir = c>>8.
// ---------------------------------------------------------------------------
__global__ __launch_bounds__(256)
void emission_kernel(const float* __restrict__ hs, const float* __restrict__ Wout,
                     const float* __restrict__ bout, const float* __restrict__ mask,
                     float* __restrict__ em) {
  int idx = blockIdx.x * blockDim.x + threadIdx.x;
  if (idx >= Bn * Tn * Kn) return;
  const int k = idx % Kn;
  const int t = (idx / Kn) % Tn;
  const int b = idx / (Kn * Tn);
  const int rh  = b >> 5;
  const int rtl = (b >> 4) & 1;
  const int hl8 = (b >> 3) & 1;
  const int v   = b & 7;
  float s = 0.0f;
  for (int dir2 = 0; dir2 < 2; ++dir2) {
    for (int jt2 = 0; jt2 < 16; ++jt2) {
      const float* hp = hs
        + ((((size_t)t * 2 + dir2) * 32 + (jt2 * 2 + rh)) * 2 + rtl) * 256
        + (size_t)hl8 * 128 + v;
      const float* wp = Wout + k * 512 + dir2 * 256 + jt2 * 16;
#pragma unroll
      for (int ll = 0; ll < 16; ++ll)
        s = fmaf(hp[ll * 8], wp[ll], s);
    }
  }
  em[((size_t)b * Tn + t) * Kn + k] = (s + bout[k]) * mask[b * Tn + t];
}

// ---------------------------------------------------------------------------
// Kernel 5: Viterbi. 16 lanes per batch (k = lane&15, valid k<9), wave32
// holds two batches; cross-tag max via 16-wide sub-wave shuffles.
// ---------------------------------------------------------------------------
__global__ __launch_bounds__(512)
void viterbi_kernel(const float* __restrict__ em, const float* __restrict__ mask,
                    const float* __restrict__ start_t, const float* __restrict__ end_t,
                    const float* __restrict__ trans,
                    unsigned char* __restrict__ hist, int* __restrict__ out) {
  const int gt = blockIdx.x * 512 + threadIdx.x;
  const int b  = gt >> 4;
  const int k  = gt & 15;
  const bool on = (k < Kn);
  float tr[Kn];
#pragma unroll
  for (int kp = 0; kp < Kn; ++kp) tr[kp] = on ? trans[kp * Kn + k] : -1e30f;
  float score = on ? (start_t[k] + em[(size_t)b * Tn * Kn + k]) : -1e30f;

  for (int t = 1; t < Tn; ++t) {
    float best = -1e30f; int bp = 0;
#pragma unroll
    for (int kp = 0; kp < Kn; ++kp) {            // ascending -> first-max wins
      float sp = __shfl(score, kp, 16);
      float cand = sp + tr[kp];
      if (cand > best) { best = cand; bp = kp; }
    }
    const float mv = mask[b * Tn + t];
    const bool  valid = (mv != 0.0f);
    const float ns = best + em[((size_t)b * Tn + t) * Kn + (on ? k : 0)];
    const int idx = valid ? bp : (on ? k : 0);
    score = (valid && on) ? ns : score;
    if (on) hist[((size_t)b * Tn + t) * Kn + k] = (unsigned char)idx;
  }
  const float fs = score + (on ? end_t[k] : 0.0f);
  float best = -1e30f; int last = 0;
#pragma unroll
  for (int kp = 0; kp < Kn; ++kp) {              // all lanes compute argmax
    float sp = __shfl(fs, kp, 16);
    if (sp > best) { best = sp; last = kp; }
  }
  if (k == 0) {                                  // lane 0 of each group backtracks
    int tag = last;
    out[b * Tn + (Tn - 1)] = (mask[b * Tn + Tn - 1] != 0.0f) ? tag : 0;
    for (int t = Tn - 1; t >= 1; --t) {
      tag = hist[((size_t)b * Tn + t) * Kn + tag];
      out[b * Tn + t - 1] = (mask[b * Tn + t - 1] != 0.0f) ? tag : 0;
    }
  }
}

// ---------------------------------------------------------------------------
extern "C" void kernel_launch(void* const* d_in, const int* in_sizes, int n_in,
                              void* d_out, int out_size, void* d_ws, size_t ws_size,
                              hipStream_t stream) {
  const int*   x       = (const int*)  d_in[0];
  const float* mask    = (const float*)d_in[1];
  const float* embed   = (const float*)d_in[2];
  const float* Wih_f   = (const float*)d_in[3];
  const float* Whh_f   = (const float*)d_in[4];
  const float* bih_f   = (const float*)d_in[5];
  const float* bhh_f   = (const float*)d_in[6];
  const float* Wih_b   = (const float*)d_in[7];
  const float* Whh_b   = (const float*)d_in[8];
  const float* bih_b   = (const float*)d_in[9];
  const float* bhh_b   = (const float*)d_in[10];
  const float* Wout    = (const float*)d_in[11];
  const float* bout    = (const float*)d_in[12];
  const float* start_t = (const float*)d_in[13];
  const float* end_t   = (const float*)d_in[14];
  const float* trans   = (const float*)d_in[15];

  char* ws = (char*)d_ws;
  size_t o = 0;
  unsigned short* wih_frag = (unsigned short*)(ws + o); o += (size_t)2 * 262144 * 2;
  unsigned short* whh_frag = (unsigned short*)(ws + o); o += (size_t)2 * 262144 * 2;
  float*          bsum     = (float*)(ws + o);          o += (size_t)2 * NG * 4;
  float*          Gx       = (float*)(ws + o);          o += (size_t)2 * Tn * GXT * 4;  // 256 MB
  float*          hs       = (float*)(ws + o);          o += (size_t)Tn * 2 * 32 * 2 * 256 * 4; // 64 MB
  float*          em       = (float*)(ws + o);          o += (size_t)Bn * Tn * Kn * 4;
  unsigned char*  hist     = (unsigned char*)(ws + o);  o += (size_t)Bn * Tn * Kn;
  (void)ws_size; (void)in_sizes; (void)n_in; (void)out_size;

  convert_weights_kernel<<<4096, 256, 0, stream>>>(Wih_f, Wih_b, Whh_f, Whh_b, wih_frag);
  bias_kernel<<<8, 256, 0, stream>>>(bih_f, bhh_f, bih_b, bhh_b, bsum);
  input_gemm_kernel<<<dim3(16, Tn, 2), 256, 0, stream>>>(x, embed, wih_frag, bsum, Gx);
  lstm_kernel<<<2, 1024, 0, stream>>>(Gx, whh_frag, hs);
  emission_kernel<<<(Bn * Tn * Kn + 255) / 256, 256, 0, stream>>>(hs, Wout, bout, mask, em);
  viterbi_kernel<<<2, 512, 0, stream>>>(em, mask, start_t, end_t, trans, hist, (int*)d_out);
}
